// LightGCN_35716948033731
// MI455X (gfx1250) — matrix operand order repaired and checked by reading, not compiled, not run on hardware
//
#include <hip/hip_runtime.h>

#define NN 16384
#define DD 64

typedef __attribute__((ext_vector_type(16))) __bf16 v16bf;
typedef __attribute__((ext_vector_type(8)))  float  v8f;

union BVec { v16bf v; uint4 q[2]; };

static __device__ __forceinline__ __bf16 f2bf(float x) { return (__bf16)x; }

// ---------------------------------------------------------------------------
// Kernel 1: rowsum of (A + I) -> d[i] = rsqrt(rowsum)   (inf -> 0 guard)
// One 256-thread block per row; float4-coalesced streaming of 64 KB/row.
// ---------------------------------------------------------------------------
__global__ __launch_bounds__(256) void rowsum_d_kernel(
    const float* __restrict__ adj, float* __restrict__ d) {
  __shared__ float red[256];
  const int row = blockIdx.x;
  const int t   = threadIdx.x;
  const float4* rowp = (const float4*)(adj + (size_t)row * NN);
  float s = 0.f;
#pragma unroll
  for (int c = 0; c < 16; ++c) {
    float4 v = rowp[c * 256 + t];
    s += (v.x + v.y) + (v.z + v.w);
  }
  red[t] = s;
  __syncthreads();
  for (int off = 128; off > 0; off >>= 1) {
    if (t < off) red[t] += red[t + off];
    __syncthreads();
  }
  if (t == 0) {
    float total = red[0] + 1.0f;  // + identity diagonal
    float r = rsqrtf(total);
    if (isinf(r)) r = 0.f;
    d[row] = r;
  }
}

// ---------------------------------------------------------------------------
// Kernel 2: pack B = bf16(d[k] * emb[k,n]) into the WMMA B-operand layout.
// Layout unit g = ((kb*4 + nt)*32 + lane): 16 bf16 (32 B) holding, for this
// lane, column n = nt*16 + (lane&15) and K = kb*32 + (lane>=16 ? 16 : 0) + e.
// Hot loop can then fetch each B tile with two contiguous b128 loads/lane.
// ---------------------------------------------------------------------------
__global__ __launch_bounds__(256) void pack_b_kernel(
    const float* __restrict__ emb, const float* __restrict__ d,
    __bf16* __restrict__ wsB) {
  const int g    = blockIdx.x * 256 + threadIdx.x;   // 512*4*32 = 65536
  const int kb   = g >> 7;
  const int rem  = g & 127;
  const int nt   = rem >> 5;
  const int lane = rem & 31;
  const int n     = nt * 16 + (lane & 15);
  const int kbase = kb * 32 + ((lane >> 4) << 4);
  BVec b;
#pragma unroll
  for (int e = 0; e < 16; ++e) {
    const int k = kbase + e;
    b.v[e] = f2bf(d[k] * emb[(size_t)k * DD + n]);
  }
  uint4* dst = (uint4*)wsB + (size_t)g * 2;
  dst[0] = b.q[0];
  dst[1] = b.q[1];
}

// ---------------------------------------------------------------------------
// Kernel 3 helpers: explicit 2-deep software pipeline.
// ---------------------------------------------------------------------------
static __device__ __forceinline__ void load_tiles(
    const float* __restrict__ arow, const uint4* __restrict__ bbase,
    int kb, int selA, int lane, float4 (&ar)[4], BVec (&b)[4]) {
  const int col = kb * 32 + selA;
  const float4* ap  = (const float4*)(arow + col);
  const float4* ap2 = (const float4*)(arow + col + 8);
  ar[0] = ap[0];  ar[1] = ap[1];
  ar[2] = ap2[0]; ar[3] = ap2[1];
  // GL2 prefetch of the A stream ~8 K-steps (1 KiB) ahead
  __builtin_prefetch((const void*)(arow + col + 256), 0, 0);
  const uint4* bp = bbase + ((size_t)kb * 128 + lane) * 2;
  b[0].q[0] = bp[0];   b[0].q[1] = bp[1];
  b[1].q[0] = bp[64];  b[1].q[1] = bp[65];
  b[2].q[0] = bp[128]; b[2].q[1] = bp[129];
  b[3].q[0] = bp[192]; b[3].q[1] = bp[193];
}

// av is per-pipeline-stage storage so the next stage's v_cvt_pk_bf16_f32
// writes different registers than the in-flight WMMAs read (kills the
// 4x v_nop WMMA->VALU WAR hazard fill).
static __device__ __forceinline__ void mma_step(
    const float4 (&ar)[4], const BVec (&b)[4], v16bf& av,
    v8f& c0, v8f& c1, v8f& c2, v8f& c3) {
  av[0]  = f2bf(ar[0].x); av[1]  = f2bf(ar[0].y); av[2]  = f2bf(ar[0].z); av[3]  = f2bf(ar[0].w);
  av[4]  = f2bf(ar[1].x); av[5]  = f2bf(ar[1].y); av[6]  = f2bf(ar[1].z); av[7]  = f2bf(ar[1].w);
  av[8]  = f2bf(ar[2].x); av[9]  = f2bf(ar[2].y); av[10] = f2bf(ar[2].z); av[11] = f2bf(ar[2].w);
  av[12] = f2bf(ar[3].x); av[13] = f2bf(ar[3].y); av[14] = f2bf(ar[3].z); av[15] = f2bf(ar[3].w);
  c0 = __builtin_amdgcn_wmma_f32_16x16x32_bf16(false, av, false, b[0].v, (short)0, c0, false, false);
  c1 = __builtin_amdgcn_wmma_f32_16x16x32_bf16(false, av, false, b[1].v, (short)0, c1, false, false);
  c2 = __builtin_amdgcn_wmma_f32_16x16x32_bf16(false, av, false, b[2].v, (short)0, c2, false, false);
  c3 = __builtin_amdgcn_wmma_f32_16x16x32_bf16(false, av, false, b[3].v, (short)0, c3, false, false);
}

// ---------------------------------------------------------------------------
// Kernel 3: out[i,:] = d[i] * ( sum_j A[i,j]*d[j]*emb[j,:] + d[i]*emb[i,:] )
// Block = 8 wave32s, one 16-row output tile per block. Each wave owns a
// disjoint K range (deterministic). Ping-pong double buffering: all 12 b128
// loads for step kb+1 issue before the 4 WMMAs of step kb.
// ---------------------------------------------------------------------------
__global__ __launch_bounds__(256) void spmm_wmma_kernel(
    const float* __restrict__ adj, const float* __restrict__ emb,
    const float* __restrict__ d, const __bf16* __restrict__ wsB,
    float* __restrict__ out) {
  __shared__ float part[8][16][DD];  // 32 KB

  const int tid     = threadIdx.x;
  const int wave    = tid >> 5;
  const int lane    = tid & 31;
  const int rowBase = blockIdx.x * 16;
  const int mlane   = lane & 15;
  const int hi      = lane >> 4;          // 0 or 1
  const int selA    = hi << 4;            // A layout: +16 K for upper half-wave

  v8f c0 = {}, c1 = {}, c2 = {}, c3 = {};

  const float* arow  = adj + (size_t)(rowBase + mlane) * NN;
  const uint4* bbase = (const uint4*)wsB;
  const int kb0 = wave * 64;              // 512 K-blocks / 8 waves (64 each, even)
  const int kb1 = kb0 + 64;

  float4 arP[4]; BVec bP[4]; v16bf avP;   // ping
  float4 arQ[4]; BVec bQ[4]; v16bf avQ;   // pong

  load_tiles(arow, bbase, kb0, selA, lane, arP, bP);
  for (int kb = kb0; kb < kb1 - 2; kb += 2) {
    load_tiles(arow, bbase, kb + 1, selA, lane, arQ, bQ);
    mma_step(arP, bP, avP, c0, c1, c2, c3);
    load_tiles(arow, bbase, kb + 2, selA, lane, arP, bP);
    mma_step(arQ, bQ, avQ, c0, c1, c2, c3);
  }
  load_tiles(arow, bbase, kb1 - 1, selA, lane, arQ, bQ);
  mma_step(arP, bP, avP, c0, c1, c2, c3);
  mma_step(arQ, bQ, avQ, c0, c1, c2, c3);

  // C/D layout: VGPR r of lane l holds C[M = r + 8*(l>=16), N = l&15]
#pragma unroll
  for (int r = 0; r < 8; ++r) {
    const int m = r + (hi << 3);
    part[wave][m][mlane +  0] = c0[r];
    part[wave][m][mlane + 16] = c1[r];
    part[wave][m][mlane + 32] = c2[r];
    part[wave][m][mlane + 48] = c3[r];
  }
  __syncthreads();

  // Reduce 8 wave-partials; apply d[i] scaling and exact fp32 diagonal term.
#pragma unroll
  for (int it = 0; it < 4; ++it) {
    const int idx = it * 256 + tid;       // 1024 outputs per block
    const int m = idx >> 6;
    const int n = idx & 63;
    float s = 0.f;
#pragma unroll
    for (int w = 0; w < 8; ++w) s += part[w][m][n];
    const int i = rowBase + m;
    const float di = d[i];
    out[(size_t)i * DD + n] = di * (s + di * emb[(size_t)i * DD + n]);
  }
}

// ---------------------------------------------------------------------------
extern "C" void kernel_launch(void* const* d_in, const int* in_sizes, int n_in,
                              void* d_out, int out_size, void* d_ws, size_t ws_size,
                              hipStream_t stream) {
  const float* adj = (const float*)d_in[0];   // [N, N] fp32
  const float* emb = (const float*)d_in[1];   // [N, D] fp32
  float* out = (float*)d_out;                 // [N, D] fp32

  float*  wsD = (float*)d_ws;                              // N floats (64 KB)
  __bf16* wsB = (__bf16*)((char*)d_ws + 64 * 1024);        // N*D bf16 (2 MiB)

  // 1) rowsum(A+I) -> d = rsqrt
  rowsum_d_kernel<<<NN, 256, 0, stream>>>(adj, wsD);
  // 2) pack B = bf16(d[k]*emb[k,:]) in WMMA B layout (65536 packing units)
  pack_b_kernel<<<256, 256, 0, stream>>>(emb, wsD, wsB);
  // 3) WMMA SpMM + diagonal + d[i] scaling
  spmm_wmma_kernel<<<NN / 16, 256, 0, stream>>>(adj, emb, wsD, wsB, out);
}
